// FieldAwareFM_81621558493775
// MI455X (gfx1250) — compile-verified, matrix operand order checked by reading, not run on hardware
//
#include <hip/hip_runtime.h>
#include <hip/hip_bf16.h>

typedef __bf16 v16bf __attribute__((ext_vector_type(16)));
typedef __bf16 v8bf  __attribute__((ext_vector_type(8)));
typedef __bf16 v2bf  __attribute__((ext_vector_type(2)));
typedef float  v8f   __attribute__((ext_vector_type(8)));
typedef float  v4f   __attribute__((ext_vector_type(4)));

#define F_DIM   1024
#define B_DIM   16384
#define E_DIM   64
#define NBLK    8       // 1024 / 128 blocks per dim
#define NPAIR   36      // NBLK*(NBLK+1)/2 upper-triangular blocks
#define KSPLIT  16
#define KLEN    (B_DIM / KSPLIT)   // 1024
#define CH      32                 // K-chunk per LDS stage
#define SK      40                 // padded bf16 row stride (80B, 16B aligned)

union FragU { v16bf v; v8bf h[2]; };

// Pack two f32 into one dword of bf16 pair -> single v_cvt_pk_bf16_f32.
static __device__ __forceinline__ unsigned pack2(float lo, float hi) {
    unsigned u;
#if __has_builtin(__builtin_amdgcn_cvt_pk_bf16_f32)
    auto v = __builtin_amdgcn_cvt_pk_bf16_f32(lo, hi);
    __builtin_memcpy(&u, &v, sizeof(u));
#else
    v2bf v;
    v[0] = (__bf16)lo;
    v[1] = (__bf16)hi;
    __builtin_memcpy(&u, &v, sizeof(u));
#endif
    return u;
}

// ---------------------------------------------------------------------------
// Kernel 1: masked FFM interaction matrix  A[i,j] = <vs[f_j,i,:], vs[f_i,j,:]>
// for i < j, else 0. Only upper-triangular 16x16 tile blocks are written.
// ---------------------------------------------------------------------------
__global__ void ffm_amask(const float* __restrict__ vs,
                          const int* __restrict__ fields,
                          float* __restrict__ A) {
    int bj = blockIdx.x, bi = blockIdx.y;
    if (bi > bj) return;
    int t = threadIdx.x;
    int i = bi * 16 + (t >> 4);
    int j = bj * 16 + (t & 15);
    float s = 0.f;
    if (i < j) {
        int fi = fields[i], fj = fields[j];
        const v4f* p = (const v4f*)(vs + ((long)fj * F_DIM + i) * E_DIM);
        const v4f* q = (const v4f*)(vs + ((long)fi * F_DIM + j) * E_DIM);
#pragma unroll
        for (int e = 0; e < E_DIM / 4; ++e) {
            v4f a = p[e], b = q[e];
            s += a[0]*b[0] + a[1]*b[1] + a[2]*b[2] + a[3]*b[3];
        }
    }
    A[i * F_DIM + j] = s;
}

// ---------------------------------------------------------------------------
// Kernel 2: fused P = X^T X (bf16 WMMA, f32 acc) with masked A-weighted
// reduction to a scalar. Each workgroup: one 128x128 block x one K-slice.
// ---------------------------------------------------------------------------
__global__ void __launch_bounds__(256)
ffm_gemm(const float* __restrict__ X,
         const float* __restrict__ Amask,
         float* __restrict__ s2) {
    __shared__ alignas(16) __bf16 As[128 * SK];
    __shared__ alignas(16) __bf16 Bs[128 * SK];
    __shared__ float wred[8];

    int bid = blockIdx.x;
    int p  = bid / KSPLIT;
    int ks = bid % KSPLIT;
    // unrank pair index -> (bi, bj) with bj >= bi
    int bi = 0, rem = p, row = NBLK;
    while (rem >= row) { rem -= row; ++bi; --row; }
    int bj = bi + rem;
    const int ibase = bi * 128, jbase = bj * 128;
    const int k0 = ks * KLEN;

    const int t    = threadIdx.x;
    const int lane = t & 31;
    const int wave = t >> 5;
    const int wr   = wave >> 1;   // 0..3 -> 32 rows each
    const int wc   = wave & 1;    // 0..1 -> 64 cols each

    // loader role: 16 k-pairs x 16 i-groups of 8
    const int k2 = (t >> 4) * 2;  // even k within chunk
    const int g  = t & 15;        // i-group

    v8f acc[2][4] = {};
    v4f a00, a01, a10, a11, b00, b01, b10, b11;

    auto load_chunk = [&](int kc) {
        const float* pa0 = X + (long)(k0 + kc + k2) * F_DIM + ibase + g * 8;
        const float* pb0 = X + (long)(k0 + kc + k2) * F_DIM + jbase + g * 8;
        a00 = *(const v4f*)(pa0);          a01 = *(const v4f*)(pa0 + 4);
        a10 = *(const v4f*)(pa0 + F_DIM);  a11 = *(const v4f*)(pa0 + F_DIM + 4);
        b00 = *(const v4f*)(pb0);          b01 = *(const v4f*)(pb0 + 4);
        b10 = *(const v4f*)(pb0 + F_DIM);  b11 = *(const v4f*)(pb0 + F_DIM + 4);
    };

    auto store_chunk = [&]() {
#pragma unroll
        for (int d = 0; d < 8; ++d) {
            float va0 = (d < 4) ? a00[d & 3] : a01[d & 3];
            float va1 = (d < 4) ? a10[d & 3] : a11[d & 3];
            float vb0 = (d < 4) ? b00[d & 3] : b01[d & 3];
            float vb1 = (d < 4) ? b10[d & 3] : b11[d & 3];
            int il = g * 8 + d;
            *(unsigned*)(&As[il * SK + k2]) = pack2(va0, va1);
            *(unsigned*)(&Bs[il * SK + k2]) = pack2(vb0, vb1);
        }
    };

    const int m    = lane & 15;
    const int half = lane >> 4;

    load_chunk(0);
    for (int kc = 0; kc < KLEN; kc += CH) {
        __syncthreads();          // previous compute done reading LDS
        store_chunk();
        __syncthreads();
        if (kc + CH < KLEN) load_chunk(kc + CH);  // prefetch next chunk

        // A/B fragments: 16-bit 16x32 layout (lanes 0-15: K0-7,K16-23;
        // lanes 16-31: K8-15,K24-31), tiles stored [i][k] in LDS.
        FragU afr[2], bfr[4];
#pragma unroll
        for (int r = 0; r < 2; ++r) {
            int base = (wr * 32 + r * 16 + m) * SK + half * 8;
            afr[r].h[0] = *(const v8bf*)(As + base);
            afr[r].h[1] = *(const v8bf*)(As + base + 16);
        }
#pragma unroll
        for (int c = 0; c < 4; ++c) {
            int base = (wc * 64 + c * 16 + m) * SK + half * 8;
            bfr[c].h[0] = *(const v8bf*)(Bs + base);
            bfr[c].h[1] = *(const v8bf*)(Bs + base + 16);
        }
#pragma unroll
        for (int r = 0; r < 2; ++r)
#pragma unroll
            for (int c = 0; c < 4; ++c)
                acc[r][c] = __builtin_amdgcn_wmma_f32_16x16x32_bf16(
                    false, afr[r].v, false, bfr[c].v,
                    (short)0, acc[r][c], false, false);
    }

    // Epilogue: sum A[i,j] * P_partial[i,j] over this block.
    float psum = 0.f;
#pragma unroll
    for (int r = 0; r < 2; ++r) {
#pragma unroll
        for (int c = 0; c < 4; ++c) {
            int i0 = ibase + wr * 32 + r * 16 + half * 8;   // + v below
            int j  = jbase + wc * 64 + c * 16 + m;
#pragma unroll
            for (int v = 0; v < 8; ++v)
                psum += acc[r][c][v] * Amask[(long)(i0 + v) * F_DIM + j];
        }
    }
#pragma unroll
    for (int off = 16; off > 0; off >>= 1)
        psum += __shfl_xor(psum, off, 32);
    if (lane == 0) wred[wave] = psum;
    __syncthreads();
    if (t == 0) {
        float s = 0.f;
#pragma unroll
        for (int w = 0; w < 8; ++w) s += wred[w];
        atomicAdd(s2, s);
    }
}

// ---------------------------------------------------------------------------
// Kernel 3: linear term, one wave32 per row.
// ---------------------------------------------------------------------------
__global__ void ffm_linear(const float* __restrict__ X,
                           const float* __restrict__ w,
                           const float* __restrict__ bias,
                           float* __restrict__ lin) {
    int wave = threadIdx.x >> 5, lane = threadIdx.x & 31;
    int b = blockIdx.x * 8 + wave;
    const v4f* xr = (const v4f*)(X + (long)b * F_DIM);
    const v4f* wr = (const v4f*)w;
    float s = 0.f;
#pragma unroll
    for (int k = 0; k < F_DIM / 128; ++k) {
        int idx = k * 32 + lane;
        v4f xv = xr[idx], wv = wr[idx];
        s += xv[0]*wv[0] + xv[1]*wv[1] + xv[2]*wv[2] + xv[3]*wv[3];
    }
#pragma unroll
    for (int off = 16; off > 0; off >>= 1)
        s += __shfl_xor(s, off, 32);
    if (lane == 0) lin[b] = s + bias[0];
}

// ---------------------------------------------------------------------------
// Kernel 4: out[b] = sigmoid(linear[b] + second_order)
// ---------------------------------------------------------------------------
__global__ void ffm_final(const float* __restrict__ lin,
                          const float* __restrict__ s2,
                          float* __restrict__ out) {
    int i = blockIdx.x * blockDim.x + threadIdx.x;
    if (i < B_DIM) {
        float x = lin[i] + s2[0];
        out[i] = 1.f / (1.f + __expf(-x));
    }
}

extern "C" void kernel_launch(void* const* d_in, const int* in_sizes, int n_in,
                              void* d_out, int out_size, void* d_ws, size_t ws_size,
                              hipStream_t stream) {
    const float* X      = (const float*)d_in[0];
    const int*   fields = (const int*)d_in[1];
    const float* w      = (const float*)d_in[2];
    const float* bias   = (const float*)d_in[3];
    const float* vs     = (const float*)d_in[4];
    float* out = (float*)d_out;

    float* Amask = (float*)d_ws;                                  // 4 MB
    float* lin   = (float*)((char*)d_ws + (4u << 20));            // 64 KB
    float* s2    = lin + B_DIM;                                   // 4 B

    hipMemsetAsync(s2, 0, sizeof(float), stream);
    ffm_amask<<<dim3(64, 64), 256, 0, stream>>>(vs, fields, Amask);
    ffm_gemm<<<dim3(NPAIR * KSPLIT), 256, 0, stream>>>(X, Amask, s2);
    ffm_linear<<<B_DIM / 8, 256, 0, stream>>>(X, w, bias, lin);
    ffm_final<<<B_DIM / 256, 256, 0, stream>>>(lin, s2, out);
}
